// FractionalAttention_39273180955144
// MI455X (gfx1250) — matrix operand order repaired and checked by reading, not compiled, not run on hardware
//
#include <hip/hip_runtime.h>
#include <hip/hip_bf16.h>
#include <math.h>

// ---------------------------------------------------------------------------
// FractionalAttention for MI455X (gfx1250): f16 WMMA flash attention.
//   B=4, S=4096, D=256, H=0.1
//   softmax(scores + log_bias): row-normalization of the fractional kernel is
//   a per-row constant -> cancels; bias = (H-1.5)*ln(|q-k|+1e-9).
//   Computed in base-2: sv2 = scores*(SCL*log2e) + (H-1.5)*log2(|q-k|+eps).
// ---------------------------------------------------------------------------

#define BB    4
#define SS    4096
#define DD    256
#define KT    32               // keys per flash iteration
#define NKT   (SS / KT)        // 128
#define QT    64               // queries per workgroup
#define C_H   (-1.4f)          // HURST - 1.5
#define SCL2  (0.09016844f)    // (1/16) * log2(e)

typedef _Float16 v16h __attribute__((ext_vector_type(16)));
typedef _Float16 v8h  __attribute__((ext_vector_type(8)));
typedef _Float16 v4h  __attribute__((ext_vector_type(4)));
typedef float    v8f  __attribute__((ext_vector_type(8)));
typedef unsigned int u32x4 __attribute__((ext_vector_type(4)));
typedef int          i32x4 __attribute__((ext_vector_type(4)));
typedef int          i32x8 __attribute__((ext_vector_type(8)));

#if defined(__has_builtin)
#  if __has_builtin(__builtin_amdgcn_tensor_load_to_lds)
#    define USE_TDM 1
#  endif
#endif
#ifndef USE_TDM
#  define USE_TDM 0
#endif

__device__ __forceinline__ v16h mk16(v8h lo, v8h hi) {
  return __builtin_shufflevector(lo, hi, 0,1,2,3,4,5,6,7,8,9,10,11,12,13,14,15);
}
// B-fragment: 16 contiguous f16 per lane (K contiguous), lane = N
__device__ __forceinline__ v16h load_bfrag(const _Float16* p) {
  return mk16(*(const v8h*)p, *(const v8h*)(p + 8));
}
// A-fragment: 16 contiguous f16 per lane (K contiguous), lane = M
__device__ __forceinline__ v16h load_afrag(const _Float16* p) {
  return mk16(*(const v8h*)p, *(const v8h*)(p + 8));
}

#if USE_TDM
// 2D tile (rows x rowlen f16, unit row stride) global -> LDS via TDM.
// D# per CDNA5 ISA: group0/group1 populated, 2/3 zero.
__device__ __forceinline__ void tdm_load_2d(unsigned int lds_off, const void* gptr,
                                            unsigned int rows, unsigned int rowlen) {
  unsigned long long ga = (unsigned long long)(uintptr_t)gptr;
  u32x4 g0;
  g0[0] = 1u;                                        // count=1, user descriptor
  g0[1] = lds_off;                                   // lds_addr (bytes)
  g0[2] = (unsigned int)ga;                          // global_addr[31:0]
  g0[3] = (unsigned int)((ga >> 32) & 0x01FFFFFFu) | (2u << 30);  // addr[56:32], type=2
  i32x8 g1;
  g1[0] = (int)(1u << 16);                           // data_size = 2B
  g1[1] = (int)((rowlen & 0xFFFFu) << 16);           // tensor_dim0[15:0]
  g1[2] = (int)(((rowlen >> 16) & 0xFFFFu) | ((rows & 0xFFFFu) << 16)); // dim0 hi, dim1 lo
  g1[3] = (int)(((rows >> 16) & 0xFFFFu) | ((rowlen & 0xFFFFu) << 16)); // dim1 hi, tile_dim0
  g1[4] = (int)(rows & 0xFFFFu);                     // tile_dim1; tile_dim2 = 0
  g1[5] = (int)rowlen;                               // tensor_dim0_stride[31:0]
  g1[6] = 0;
  g1[7] = 0;
  i32x4 z4 = {0, 0, 0, 0};
#if __clang_major__ >= 23
  i32x8 z8 = {0, 0, 0, 0, 0, 0, 0, 0};
  __builtin_amdgcn_tensor_load_to_lds(g0, g1, z4, z4, z8, 0);
#else
  __builtin_amdgcn_tensor_load_to_lds(g0, g1, z4, z4, 0);
#endif
}
#endif

// ---------------------------------------------------------------------------
// f32 -> f16 convert (vectorized; all sizes are multiples of 4)
// ---------------------------------------------------------------------------
__global__ void fa_cvt_f16(const float* __restrict__ src, _Float16* __restrict__ dst, int n) {
  int i4 = (blockIdx.x * blockDim.x + threadIdx.x) * 4;
  if (i4 + 3 < n) {
    float4 f = *(const float4*)(src + i4);
    v4h h = {(_Float16)f.x, (_Float16)f.y, (_Float16)f.z, (_Float16)f.w};
    *(v4h*)(dst + i4) = h;
  }
}

// ---------------------------------------------------------------------------
// QKV projection:  y[s,e] = sum_d x[s,d] * W[e,d] + b[e]   (torch Linear)
// grid.x = B*S/64, grid.y = 3 (q,k,v); block = 256 (8 waves)
// wave (rgrp, chalf): rows [rgrp*16,+16), cols [chalf*128,+128)
// B fragments fully preloaded per subtile -> loads clause ahead of wmma chain.
// ---------------------------------------------------------------------------
__global__ void __launch_bounds__(256)
fa_qkv_proj(const _Float16* __restrict__ Xh, const _Float16* __restrict__ Wh,
            const float* __restrict__ bq, const float* __restrict__ bk,
            const float* __restrict__ bv,
            _Float16* __restrict__ Qh, _Float16* __restrict__ Kh, _Float16* __restrict__ Vh) {
  const int tid   = threadIdx.x;
  const int lane  = tid & 31;
  const int wave  = tid >> 5;
  const int rgrp  = wave & 3;
  const int chalf = wave >> 2;
  const int row0  = blockIdx.x * 64 + rgrp * 16;
  const int n0    = chalf * 128;

  const _Float16* W    = Wh + (size_t)blockIdx.y * DD * DD;
  const float*    bias = (blockIdx.y == 0) ? bq : (blockIdx.y == 1) ? bk : bv;
  _Float16*       Out  = (blockIdx.y == 0) ? Qh : (blockIdx.y == 1) ? Kh : Vh;

  const _Float16* xr = Xh + (size_t)(row0 + (lane & 15)) * DD + ((lane & 16) ? 16 : 0);
  v16h af[8];
#pragma unroll
  for (int t = 0; t < 8; ++t) af[t] = load_afrag(xr + t * 32);

  v8f acc[8];
#pragma unroll
  for (int s = 0; s < 8; ++s) acc[s] = (v8f){0.f,0.f,0.f,0.f,0.f,0.f,0.f,0.f};

#pragma unroll
  for (int s = 0; s < 8; ++s) {
    const _Float16* wr = W + (size_t)(n0 + s * 16 + (lane & 15)) * DD + ((lane & 16) ? 16 : 0);
    v16h bf[8];
#pragma unroll
    for (int t = 0; t < 8; ++t) bf[t] = load_bfrag(wr + t * 32);
#pragma unroll
    for (int t = 0; t < 8; ++t)
      acc[s] = __builtin_amdgcn_wmma_f32_16x16x32_f16(false, af[t], false, bf[t],
                                                      (short)0, acc[s], false, false);
  }

#pragma unroll
  for (int s = 0; s < 8; ++s) {
    int col = n0 + s * 16 + (lane & 15);
    float bv2 = bias[col];
#pragma unroll
    for (int r = 0; r < 8; ++r) {
      int row = row0 + r + ((lane & 16) ? 8 : 0);
      Out[(size_t)row * DD + col] = (_Float16)(acc[s][r] + bv2);
    }
  }
}

// ---------------------------------------------------------------------------
// Fused flash attention. grid = (S/64, B), block = 256 (8 waves).
// wave (rgrp, chalf): 16 query rows (rgrp), 128 output cols (chalf).
// Double-buffered 32-key tiles: TDM for tile kt+1 issued at the top of
// iteration kt and waited just before the closing barrier -> DMA overlaps
// the 24 WMMAs + softmax of the current tile.
// ---------------------------------------------------------------------------
__global__ void __launch_bounds__(256)
fa_flash(const _Float16* __restrict__ Qh, const _Float16* __restrict__ Kh,
         const _Float16* __restrict__ Vh, float* __restrict__ out) {
  __shared__ _Float16 Kt[2][KT * DD];     // [key][d]   2 x 16 KB
  __shared__ _Float16 Vt[2][DD * KT];     // [d][key]   2 x 16 KB (transposed)
  __shared__ _Float16 Pl[4][16 * KT];     // per row-group P   4 KB

  const int tid   = threadIdx.x;
  const int lane  = tid & 31;
  const int wave  = tid >> 5;
  const int rgrp  = wave & 3;
  const int chalf = wave >> 2;
  const int b     = blockIdx.y;
  const int q0    = blockIdx.x * QT + rgrp * 16;

  const _Float16* Kg = Kh + (size_t)b * SS * DD;
  const _Float16* Vg = Vh + (size_t)b * SS * DD;

  // Q A-fragments resident in registers for the whole pass
  const _Float16* qr = Qh + (size_t)((size_t)b * SS + q0 + (lane & 15)) * DD +
                       ((lane & 16) ? 16 : 0);
  v16h qf[8];
#pragma unroll
  for (int t = 0; t < 8; ++t) qf[t] = load_afrag(qr + t * 32);

  v8f oacc[8];
#pragma unroll
  for (int n = 0; n < 8; ++n) oacc[n] = (v8f){0.f,0.f,0.f,0.f,0.f,0.f,0.f,0.f};
  float m_run[8], l_run[8];
#pragma unroll
  for (int r = 0; r < 8; ++r) { m_run[r] = -INFINITY; l_run[r] = 0.f; }

  // ---- prologue: stage tile 0 into buffer 0
  {
#if USE_TDM
    if (tid < 32) tdm_load_2d((unsigned int)(uintptr_t)&Kt[0][0], (const void*)Kg, KT, DD);
#else
    for (int i = tid; i < (KT * DD) / 8; i += 256)
      *(v8h*)&Kt[0][i * 8] = *(const v8h*)(Kg + (size_t)i * 8);
#endif
    for (int i = tid; i < (KT * DD) / 8; i += 256) {
      int key = i >> 5, d0 = (i & 31) * 8;
      v8h src = *(const v8h*)(Vg + (size_t)key * DD + d0);
#pragma unroll
      for (int j = 0; j < 8; ++j) Vt[0][(d0 + j) * KT + key] = src[j];
    }
#if USE_TDM
    if (tid < 32) __builtin_amdgcn_s_wait_tensorcnt(0);
#endif
    __syncthreads();
  }

  for (int kt = 0; kt < NKT; ++kt) {
    const int cur = kt & 1, nxt = cur ^ 1;

    // ---- stage tile kt+1 into the other buffer (overlaps with compute)
    if (kt + 1 < NKT) {
      const _Float16* kn = Kg + (size_t)(kt + 1) * KT * DD;
      const _Float16* vn = Vg + (size_t)(kt + 1) * KT * DD;
#if USE_TDM
      if (tid < 32) tdm_load_2d((unsigned int)(uintptr_t)&Kt[nxt][0], (const void*)kn, KT, DD);
#else
      for (int i = tid; i < (KT * DD) / 8; i += 256)
        *(v8h*)&Kt[nxt][i * 8] = *(const v8h*)(kn + (size_t)i * 8);
#endif
      __builtin_prefetch(vn + (size_t)tid * 32, 0, 1);
      for (int i = tid; i < (KT * DD) / 8; i += 256) {
        int key = i >> 5, d0 = (i & 31) * 8;
        v8h src = *(const v8h*)(vn + (size_t)key * DD + d0);
#pragma unroll
        for (int j = 0; j < 8; ++j) Vt[nxt][(d0 + j) * KT + key] = src[j];
      }
    }

    // ---- scores: S = Q * K^T  (16 q-rows x 32 keys), B-frags preloaded in 2 groups
    v8f sacc[2];
#pragma unroll
    for (int s = 0; s < 2; ++s) {
      sacc[s] = (v8f){0.f,0.f,0.f,0.f,0.f,0.f,0.f,0.f};
      const _Float16* kr = &Kt[cur][(s * 16 + (lane & 15)) * DD + ((lane & 16) ? 16 : 0)];
#pragma unroll
      for (int g = 0; g < 2; ++g) {
        v16h bf[4];
#pragma unroll
        for (int t = 0; t < 4; ++t) bf[t] = load_bfrag(kr + (g * 4 + t) * 32);
#pragma unroll
        for (int t = 0; t < 4; ++t)
          sacc[s] = __builtin_amdgcn_wmma_f32_16x16x32_f16(false, qf[g * 4 + t], false, bf[t],
                                                           (short)0, sacc[s], false, false);
      }
    }

    // ---- bias + online softmax in base-2 (wave32 16-lane shuffle reductions)
    float lane_k = (float)(kt * KT + (lane & 15));
    v8f ps[2], corrv;
#pragma unroll
    for (int r = 0; r < 8; ++r) {
      float qi = (float)(q0 + r + ((lane & 16) ? 8 : 0));
      float sv0 = sacc[0][r] * SCL2 + C_H * log2f(fabsf(qi - lane_k) + 1e-9f);
      float sv1 = sacc[1][r] * SCL2 + C_H * log2f(fabsf(qi - lane_k - 16.f) + 1e-9f);
      float mp  = fmaxf(sv0, sv1);
      mp = fmaxf(mp, __shfl_xor(mp, 1));
      mp = fmaxf(mp, __shfl_xor(mp, 2));
      mp = fmaxf(mp, __shfl_xor(mp, 4));
      mp = fmaxf(mp, __shfl_xor(mp, 8));
      float mnew = fmaxf(m_run[r], mp);
      float corr = exp2f(m_run[r] - mnew);
      float p0 = exp2f(sv0 - mnew);
      float p1 = exp2f(sv1 - mnew);
      ps[0][r] = p0;
      ps[1][r] = p1;
      float rs = p0 + p1;
      rs += __shfl_xor(rs, 1);
      rs += __shfl_xor(rs, 2);
      rs += __shfl_xor(rs, 4);
      rs += __shfl_xor(rs, 8);
      l_run[r] = l_run[r] * corr + rs;
      m_run[r] = mnew;
      corrv[r] = corr;
    }
#pragma unroll
    for (int n = 0; n < 8; ++n) oacc[n] *= corrv;

    // ---- P -> LDS (one d-half writes), A-fragment reload after barrier
    if (chalf == 0) {
#pragma unroll
      for (int s = 0; s < 2; ++s)
#pragma unroll
        for (int r = 0; r < 8; ++r) {
          int qm = r + ((lane & 16) ? 8 : 0);
          Pl[rgrp][qm * KT + s * 16 + (lane & 15)] = (_Float16)ps[s][r];
        }
    }
    __syncthreads();   // sync1: Pl visible (Kt[cur] reads complete)

    const _Float16* pr = &Pl[rgrp][(lane & 15) * KT + ((lane & 16) ? 16 : 0)];
    v16h pa = load_afrag(pr);

    // ---- O += P * V  (K = 32 -> one wmma per d-subtile), V-frags in 2 groups
#pragma unroll
    for (int g = 0; g < 2; ++g) {
      v16h vf[4];
#pragma unroll
      for (int n = 0; n < 4; ++n)
        vf[n] = load_bfrag(&Vt[cur][(chalf * 128 + (g * 4 + n) * 16 + (lane & 15)) * KT +
                                    ((lane & 16) ? 16 : 0)]);
#pragma unroll
      for (int n = 0; n < 4; ++n)
        oacc[g * 4 + n] = __builtin_amdgcn_wmma_f32_16x16x32_f16(false, pa, false, vf[n],
                                                                 (short)0, oacc[g * 4 + n],
                                                                 false, false);
    }

#if USE_TDM
    if (tid < 32 && kt + 1 < NKT) __builtin_amdgcn_s_wait_tensorcnt(0);
#endif
    __syncthreads();   // sync2: next buffers staged, cur buffers free
  }

  // ---- epilogue: normalize and store f32
  v8f inv;
#pragma unroll
  for (int r = 0; r < 8; ++r) inv[r] = 1.0f / l_run[r];
#pragma unroll
  for (int n = 0; n < 8; ++n) {
    int col = chalf * 128 + n * 16 + (lane & 15);
#pragma unroll
    for (int r = 0; r < 8; ++r) {
      int row = q0 + r + ((lane & 16) ? 8 : 0);
      out[((size_t)b * SS + row) * DD + col] = oacc[n][r] * inv[r];
    }
  }
}

// ---------------------------------------------------------------------------
// Host launcher. Workspace: Xh(8MB) | Wh(384KB) | Qh(8MB) | Kh(8MB) | Vh(8MB)
// ---------------------------------------------------------------------------
extern "C" void kernel_launch(void* const* d_in, const int* in_sizes, int n_in,
                              void* d_out, int out_size, void* d_ws, size_t ws_size,
                              hipStream_t stream) {
  (void)in_sizes; (void)n_in; (void)out_size; (void)ws_size;
  const float* x  = (const float*)d_in[0];
  const float* Wq = (const float*)d_in[1];
  const float* bq = (const float*)d_in[2];
  const float* Wk = (const float*)d_in[3];
  const float* bk = (const float*)d_in[4];
  const float* Wv = (const float*)d_in[5];
  const float* bv = (const float*)d_in[6];

  const size_t nX = (size_t)BB * SS * DD;   // 4,194,304
  const size_t nW = (size_t)DD * DD;        // 65,536

  char* w = (char*)d_ws;
  _Float16* Xh = (_Float16*)w; w += nX * sizeof(_Float16);
  _Float16* Wh = (_Float16*)w; w += 3 * nW * sizeof(_Float16);
  _Float16* Qh = (_Float16*)w; w += nX * sizeof(_Float16);
  _Float16* Kh = (_Float16*)w; w += nX * sizeof(_Float16);
  _Float16* Vh = (_Float16*)w; w += nX * sizeof(_Float16);

  fa_cvt_f16<<<(int)((nX / 4 + 255) / 256), 256, 0, stream>>>(x, Xh, (int)nX);
  fa_cvt_f16<<<(int)((nW / 4 + 255) / 256), 256, 0, stream>>>(Wq, Wh,          (int)nW);
  fa_cvt_f16<<<(int)((nW / 4 + 255) / 256), 256, 0, stream>>>(Wk, Wh + nW,     (int)nW);
  fa_cvt_f16<<<(int)((nW / 4 + 255) / 256), 256, 0, stream>>>(Wv, Wh + 2 * nW, (int)nW);

  fa_qkv_proj<<<dim3((BB * SS) / 64, 3), 256, 0, stream>>>(Xh, Wh, bq, bk, bv, Qh, Kh, Vh);
  fa_flash<<<dim3(SS / QT, BB), 256, 0, stream>>>(Qh, Kh, Vh, (float*)d_out);
}
